// Project3DTo2D_54623394071299
// MI455X (gfx1250) — compile-verified, matrix operand order checked by reading, not run on hardware
//
#include <hip/hip_runtime.h>
#include <hip/hip_bf16.h>

typedef __attribute__((ext_vector_type(16))) _Float16 v16h;
typedef __attribute__((ext_vector_type(8)))  _Float16 v8h;
typedef __attribute__((ext_vector_type(8)))  float    v8f;

#define RES    256
#define HW     (RES * RES)
#define NVIEW  3
#define KCLOUD 8
#define NPTS   200000

// normalized 5-tap gaussian, sigma = 1.5
__constant__ float G5[5] = {0.1200777f, 0.2338813f, 0.2920824f, 0.2338813f, 0.1200777f};

// Per-lane element -> K index map for 16-bit A/B fragments of 16x16x32 WMMA
// (ISA 7.12.2): element i in 0..7  -> K = 8*half + i
//               element i in 8..15 -> K = 16 + 8*half + (i-8)
__device__ __forceinline__ int kmap(int half, int i) {
  return (i < 8) ? (8 * half + i) : (16 + 8 * half + (i - 8));
}

__device__ __forceinline__ int iclamp(int x, int lo, int hi) {
  return x < lo ? lo : (x > hi ? hi : x);
}

// ---------------------------------------------------------------- stats ----
__global__ __launch_bounds__(256) void stats_kernel(const float* __restrict__ pts,
                                                    float* __restrict__ stats) {
  const int k = blockIdx.x;
  const int tid = threadIdx.x;
  const float* p = pts + (size_t)k * NPTS * 3;
  __shared__ float rx[256], ry[256], rz[256];

  float sx = 0.f, sy = 0.f, sz = 0.f;
  for (int i = tid; i < NPTS; i += 256) {
    if (i + 768 < NPTS) __builtin_prefetch(&p[3 * (i + 768)], 0, 1);  // global_prefetch_b8
    sx += p[3 * i + 0]; sy += p[3 * i + 1]; sz += p[3 * i + 2];
  }
  rx[tid] = sx; ry[tid] = sy; rz[tid] = sz;
  __syncthreads();
  for (int s = 128; s > 0; s >>= 1) {
    if (tid < s) { rx[tid] += rx[tid + s]; ry[tid] += ry[tid + s]; rz[tid] += rz[tid + s]; }
    __syncthreads();
  }
  const float mx = rx[0] * (1.0f / NPTS);
  const float my = ry[0] * (1.0f / NPTS);
  const float mz = rz[0] * (1.0f / NPTS);
  __syncthreads();

  float mm = 0.f;
  for (int i = tid; i < NPTS; i += 256) {
    mm = fmaxf(mm, fabsf(p[3 * i + 0] - mx));
    mm = fmaxf(mm, fabsf(p[3 * i + 1] - my));
    mm = fmaxf(mm, fabsf(p[3 * i + 2] - mz));
  }
  rx[tid] = mm;
  __syncthreads();
  for (int s = 128; s > 0; s >>= 1) {
    if (tid < s) rx[tid] = fmaxf(rx[tid], rx[tid + s]);
    __syncthreads();
  }
  if (tid == 0) {
    stats[4 * k + 0] = mx; stats[4 * k + 1] = my; stats[4 * k + 2] = mz;
    stats[4 * k + 3] = fmaxf(rx[0], 1e-6f);
  }
}

// ----------------------------------------------------------------- zero ----
__global__ __launch_bounds__(256) void zero_kernel(float4* __restrict__ p, size_t n4) {
  const size_t i = (size_t)blockIdx.x * 256 + threadIdx.x;
  if (i < n4) p[i] = make_float4(0.f, 0.f, 0.f, 0.f);
}

// -------------------------------------------------------------- scatter ----
__global__ __launch_bounds__(256) void scatter_kernel(
    const float* __restrict__ pts, const float* __restrict__ normals,
    const float* __restrict__ geo, const float* __restrict__ stats,
    float* __restrict__ acc, float* __restrict__ corners_out,
    float* __restrict__ weights_out) {
  const int gid = blockIdx.x * 256 + threadIdx.x;
  if (gid >= KCLOUD * NPTS) return;
  const int k = gid / NPTS;
  const float* st = stats + 4 * k;
  const float inv = 0.95f / st[3];

  float pn[3], nr[3], gf[3];
#pragma unroll
  for (int a = 0; a < 3; ++a) {
    pn[a] = (pts[(size_t)gid * 3 + a] - st[a]) * inv;
    nr[a] = normals[(size_t)gid * 3 + a];
    gf[a] = geo[(size_t)gid * 3 + a];
  }
  float* accK = acc + (size_t)k * NVIEW * HW * 8;
  const int axu[3] = {0, 2, 0}, axv[3] = {1, 1, 2}, axd[3] = {2, 0, 1};
#pragma unroll
  for (int v = 0; v < NVIEW; ++v) {
    const float uf  = (pn[axu[v]] + 0.95f) * (255.0f / 1.9f);
    const float vf  = (pn[axv[v]] + 0.95f) * (255.0f / 1.9f);
    const float dep = (pn[axd[v]] + 0.95f) * (1.0f / 1.9f);
    int u0 = iclamp((int)floorf(uf), 0, RES - 2);
    int v0 = iclamp((int)floorf(vf), 0, RES - 2);
    const float wu1 = fminf(fmaxf(uf - (float)u0, 0.f), 1.f), wu0 = 1.f - wu1;
    const float wv1 = fminf(fmaxf(vf - (float)v0, 0.f), 1.f), wv0 = 1.f - wv1;
    const int   c4[4] = {v0 * RES + u0, v0 * RES + u0 + 1,
                         (v0 + 1) * RES + u0, (v0 + 1) * RES + u0 + 1};
    const float w4[4] = {wv0 * wu0, wv0 * wu1, wv1 * wu0, wv1 * wu1};
    const size_t ob = ((size_t)gid * NVIEW + v) * 4;
#pragma unroll
    for (int j = 0; j < 4; ++j) {
      corners_out[ob + j] = (float)c4[j];
      weights_out[ob + j] = w4[j];
      float* cell = accK + ((size_t)v * HW + c4[j]) * 8;
      const float w = w4[j];
      atomicAdd(cell + 0, w);
      atomicAdd(cell + 1, dep * w);
      atomicAdd(cell + 2, nr[0] * w);
      atomicAdd(cell + 3, nr[1] * w);
      atomicAdd(cell + 4, nr[2] * w);
      atomicAdd(cell + 5, gf[0] * w);
      atomicAdd(cell + 6, gf[1] * w);
      atomicAdd(cell + 7, gf[2] * w);
    }
  }
}

// ------------------------------------------------------------ normalize ----
__global__ __launch_bounds__(256) void normalize_kernel(const float* __restrict__ acc,
                                                        float* __restrict__ pre,
                                                        float* __restrict__ cnt_out) {
  const int gid = blockIdx.x * 256 + threadIdx.x;
  if (gid >= KCLOUD * NVIEW * HW) return;
  const float* a = acc + (size_t)gid * 8;
  const float cnt  = a[0];
  const float inv  = 1.0f / fmaxf(cnt, 1e-6f);
  const float mask = cnt > 0.f ? 1.f : 0.f;
  const int pix = gid & (HW - 1);
  const int kv  = gid >> 16;  // HW == 65536
  float* o = pre + (size_t)kv * 8 * HW + pix;
  o[0 * (size_t)HW] = a[1] * inv;  // depth
  o[1 * (size_t)HW] = mask;        // mask
#pragma unroll
  for (int c = 2; c < 8; ++c) o[(size_t)c * HW] = a[c] * inv;
  cnt_out[gid] = cnt;
}

// ------------------------------------------- separable 5x5 blur via WMMA ----
// out[16x16] = Av(16x32) x [ In(32x32) x Bh(32x16) ]   (banded gaussian matmuls)
// Staging is fully 128-bit and the halo load is branchless (clamped addresses
// + v_cndmask zeroing instead of exec-masked load blocks).
__global__ __launch_bounds__(32) void gauss_wmma_kernel(const float* __restrict__ pre,
                                                        float* __restrict__ out) {
  __shared__ _Float16 lin[32][32];    // input tile + halo (rows/cols >=20 zero)
  __shared__ _Float16 ltmp[16][32];   // h-pass result, TRANSPOSED: [col][row]
  const int img  = blockIdx.y;                // 0..K*V*8-1
  const int tr   = (blockIdx.x >> 4) << 4;    // tile row origin
  const int tc   = (blockIdx.x & 15) << 4;    // tile col origin
  const int lane = threadIdx.x;
  const float* src = pre + (size_t)img * HW;

  // stage 20x20 halo tile (zero-padded) as f16; unconditional clamped loads,
  // zeroed by select; 4x ds_store_b128 per lane
  const int  r      = tr - 2 + lane;
  const bool rvalid = (lane < 20) && (r >= 0) && (r < RES);
  const float* rowp = src + (size_t)iclamp(r, 0, RES - 1) * RES;
#pragma unroll
  for (int q = 0; q < 4; ++q) {
    v8h h;
#pragma unroll
    for (int e = 0; e < 8; ++e) {
      const int c  = 8 * q + e;          // compile-time after unroll
      const int cc = tc - 2 + c;
      float v = 0.0f;
      if (c < 20) {                      // static predicate: 20 real loads
        const float lv = rowp[iclamp(cc, 0, RES - 1)];
        v = (rvalid && cc >= 0 && cc < RES) ? lv : 0.0f;
      }
      h[e] = (_Float16)v;
    }
    *(v8h*)&lin[lane][8 * q] = h;
  }
  __syncthreads();

  const int m    = lane & 15;
  const int half = lane >> 4;

  // banded gaussian fragment: value at (row/col m, K=kk) is g[kk-m]
  v16h gband;
#pragma unroll
  for (int i = 0; i < 16; ++i) {
    const int kk = kmap(half, i);
    const int d  = kk - m;
    gband[i] = (_Float16)((d >= 0 && d <= 4) ? G5[d] : 0.0f);
  }

  // A fragments of the input tile: two contiguous 8-half runs each (b128 loads)
  v8h a0lo = *(const v8h*)&lin[m][8 * half];
  v8h a0hi = *(const v8h*)&lin[m][16 + 8 * half];
  v8h a1lo = *(const v8h*)&lin[16 + m][8 * half];
  v8h a1hi = *(const v8h*)&lin[16 + m][16 + 8 * half];
  v16h a0 = __builtin_shufflevector(a0lo, a0hi, 0,1,2,3,4,5,6,7,8,9,10,11,12,13,14,15);
  v16h a1 = __builtin_shufflevector(a1lo, a1hi, 0,1,2,3,4,5,6,7,8,9,10,11,12,13,14,15);

  v8f z = {};
  // horizontal pass: tmp = In x Bh
  v8f t0 = __builtin_amdgcn_wmma_f32_16x16x32_f16(false, a0, false, gband, (short)0, z, false, false);
  v8f t1 = __builtin_amdgcn_wmma_f32_16x16x32_f16(false, a1, false, gband, (short)0, z, false, false);

  // D layout: vgpr rr -> (row rr + 8*half, col m). Transposed store makes the
  // 8 rows contiguous: 2x ds_store_b128 per result.
  const int mofs = half * 8;
  v8h p0, p1;
#pragma unroll
  for (int e = 0; e < 8; ++e) { p0[e] = (_Float16)t0[e]; p1[e] = (_Float16)t1[e]; }
  *(v8h*)&ltmp[m][mofs]      = p0;   // tmp rows mofs..mofs+7       (col m)
  *(v8h*)&ltmp[m][16 + mofs] = p1;   // tmp rows 16+mofs..16+mofs+7 (col m)
  __syncthreads();

  // vertical pass: out = Av x tmp ; B fragment = 2x ds_load_b128
  v8h blo = *(const v8h*)&ltmp[m][8 * half];
  v8h bhi = *(const v8h*)&ltmp[m][16 + 8 * half];
  v16h bt = __builtin_shufflevector(blo, bhi, 0,1,2,3,4,5,6,7,8,9,10,11,12,13,14,15);
  v8f o = __builtin_amdgcn_wmma_f32_16x16x32_f16(false, gband, false, bt, (short)0, z, false, false);

  float* dst = out + (size_t)img * HW;
#pragma unroll
  for (int rr = 0; rr < 8; ++rr)
    dst[(size_t)(tr + rr + mofs) * RES + (tc + m)] = o[rr];
}

// ---------------------------------------------------------------- launch ----
extern "C" void kernel_launch(void* const* d_in, const int* in_sizes, int n_in,
                              void* d_out, int out_size, void* d_ws, size_t ws_size,
                              hipStream_t stream) {
  const float* pts     = (const float*)d_in[0];
  const float* normals = (const float*)d_in[1];
  const float* geo     = (const float*)d_in[2];
  float* out = (float*)d_out;

  const size_t IMGS = (size_t)KCLOUD * NVIEW * 8 * HW;    // 12,582,912
  const size_t CORN = (size_t)KCLOUD * NPTS * NVIEW * 4;  // 19,200,000

  float* imgs_out    = out;
  float* corners_out = out + IMGS;
  float* weights_out = corners_out + CORN;
  float* cnt_out     = weights_out + CORN;

  float* ws    = (float*)d_ws;
  float* stats = ws;           // 32 floats (16B-aligned tail follows)
  float* acc   = ws + 32;      // K*V*HW*8 = IMGS floats
  float* pre   = acc + IMGS;   // IMGS floats

  stats_kernel<<<KCLOUD, 256, 0, stream>>>(pts, stats);

  const size_t n4 = IMGS / 4;
  zero_kernel<<<(int)((n4 + 255) / 256), 256, 0, stream>>>((float4*)acc, n4);

  const int npts_total = KCLOUD * NPTS;
  scatter_kernel<<<(npts_total + 255) / 256, 256, 0, stream>>>(
      pts, normals, geo, stats, acc, corners_out, weights_out);

  const int npix = KCLOUD * NVIEW * HW;
  normalize_kernel<<<(npix + 255) / 256, 256, 0, stream>>>(acc, pre, cnt_out);

  dim3 grid(256, KCLOUD * NVIEW * 8);  // 256 tiles x 192 images
  gauss_wmma_kernel<<<grid, 32, 0, stream>>>(pre, imgs_out);
}
